// BitwiseTasNet_77017353552489
// MI455X (gfx1250) — matrix operand order repaired and checked by reading, not compiled, not run on hardware
//
#include <hip/hip_runtime.h>
#include <hip/hip_bf16.h>
#include <math.h>

typedef __attribute__((ext_vector_type(16))) _Float16 v16h;
typedef __attribute__((ext_vector_type(8)))  _Float16 v8h;
typedef __attribute__((ext_vector_type(8)))  float    v8f;

#define EPSBN 1e-5f
#define GTPB 256
#define MT 128
#define NT 64
#define KT 32
#define LROW 48   // padded LDS row length in halfs (96B rows -> 16B/32B aligned accesses)

// ---------------------------------------------------------------- encoder
__global__ __launch_bounds__(256) void encoder_kernel(
    const float* __restrict__ x, const float* __restrict__ ew,
    const float* __restrict__ eb, float* __restrict__ enc,
    int T, int Tc, int E, int FK, int stride, int pad)
{
  int t = blockIdx.x * blockDim.x + threadIdx.x;
  int e = blockIdx.y;
  int b = blockIdx.z;
  if (t >= Tc) return;
  const float* xb = x + (size_t)b * T;
  const float* w  = ew + (size_t)e * FK;
  float acc = eb[e];
  int base = t * stride - pad;
#pragma unroll
  for (int k = 0; k < 20; ++k) {
    int xi = base + k;
    float xv = (xi >= 0 && xi < T) ? xb[xi] : 0.f;
    acc = fmaf(xv, w[k], acc);
  }
  enc[((size_t)b * E + e) * Tc + t] = acc;
}

// ------------------------------------------------- fused WMMA GEMM (1x1 conv)
// O[b,m,t] = epilogue( sum_k W[m,k] * H[b,k,t] + bias[m] )
// mode 0: PReLU(alpha) then BN(scale,shift)   mode 1: (+ residual if given)
__global__ __launch_bounds__(GTPB) void gemm_fused_kernel(
    const float* __restrict__ W,     // [M,K] row-major
    const float* __restrict__ H,     // [B,K,Tc]
    float* __restrict__ O,           // [B,M,Tc]
    const float* __restrict__ bias,  // [M]
    const float* __restrict__ resid, // [B,M,Tc] or nullptr
    const float* __restrict__ alpha_p, // scalar PReLU slope or nullptr
    const float* __restrict__ bng, const float* __restrict__ bnb,
    const float* __restrict__ bnm, const float* __restrict__ bnv,
    int M, int K, int Tc, int mode)
{
  __shared__ __align__(32) _Float16 Alds[MT][LROW];
  __shared__ __align__(32) _Float16 Blds[NT][LROW];

  const int tid  = threadIdx.x;
  const int wv   = tid >> 5;    // wave id 0..7
  const int lane = tid & 31;
  const int lh   = lane >> 4;   // lane half
  const int ln   = lane & 15;
  const int mBase = blockIdx.y * MT;
  const int tBase = blockIdx.x * NT;
  const int b     = blockIdx.z;

  const float* Hb = H + (size_t)b * K * Tc;
  float*       Ob = O + (size_t)b * M * Tc;

  // Per-thread B-gather coordinates: one time-column, one 8-deep K strip.
  const int bn  = tid & 63;     // column within N tile
  const int bkq = tid >> 6;     // which K-strip of 8 (0..3)
  const int bt  = tBase + bn;   // global time index (same for all 8 loads)
  const bool bok = (bt < Tc);

  v8f acc[4];
#pragma unroll
  for (int i = 0; i < 4; ++i) acc[i] = (v8f){};

  for (int k0 = 0; k0 < K; k0 += KT) {
    // ---- stage A tile (128x32 f32 -> f16, contiguous k layout)
    {
      int row = tid >> 1;
      int kh  = (tid & 1) * 16;
      const float* src = W + (size_t)(mBase + row) * K + (k0 + kh);
#pragma unroll
      for (int j = 0; j < 16; ++j) {
        int k = kh + j;
        int p = k;
        Alds[row][p] = (_Float16)src[j];
      }
    }
    // ---- stage B tile (32x64): per-thread K-strip gather, transpose in regs.
    // Lanes have consecutive n -> every load is a coalesced 128B burst.
    {
      const float* src = Hb + (size_t)(k0 + bkq * 8) * Tc + bt;
      v8h strip;
      if (bok) {
#pragma unroll
        for (int j = 0; j < 8; ++j) strip[j] = (_Float16)src[(size_t)j * Tc];
        if (k0 + KT < K)  // warm L2 for next K-step of this strip
          __builtin_prefetch(src + (size_t)KT * Tc, 0, 1);
      } else {
#pragma unroll
        for (int j = 0; j < 8; ++j) strip[j] = (_Float16)0.f;
      }
      *(v8h*)&Blds[bn][bkq * 8] = strip;   // one aligned 16B LDS store
    }
    __syncthreads();

    v16h af = *(const v16h*)&Alds[wv * 16 + ln][lh * 16];
#pragma unroll
    for (int ns = 0; ns < 4; ++ns) {
      v16h bf = *(const v16h*)&Blds[ns * 16 + ln][lh * 16];
      acc[ns] = __builtin_amdgcn_wmma_f32_16x16x32_f16(
          false, af, false, bf, (short)0, acc[ns], false, false);
    }
    __syncthreads();
  }

  const float aslope = alpha_p ? alpha_p[0] : 0.f;
  const float* residB = resid ? (resid + (size_t)b * M * Tc) : nullptr;

  // Pre-compute per-row epilogue params (8 rows per lane).
  float bv_r[8], sc_r[8], sh_r[8];
#pragma unroll
  for (int r = 0; r < 8; ++r) {
    int m = mBase + wv * 16 + r + 8 * lh;   // C layout: VGPR r -> M = r + 8*lanehalf
    bv_r[r] = bias[m];
    if (mode == 0) {
      float s = bng[m] * rsqrtf(bnv[m] + EPSBN);
      sc_r[r] = s;
      sh_r[r] = bnb[m] - bnm[m] * s;
    } else {
      sc_r[r] = 1.f;
      sh_r[r] = 0.f;
    }
  }

#pragma unroll
  for (int ns = 0; ns < 4; ++ns) {
    int t = tBase + ns * 16 + ln;
    if (t >= Tc) continue;      // one guard per column group
#pragma unroll
    for (int r = 0; r < 8; ++r) {
      int m = mBase + wv * 16 + r + 8 * lh;
      float v = acc[ns][r] + bv_r[r];
      if (mode == 0) {
        v = (v > 0.f) ? v : aslope * v;
        v = fmaf(v, sc_r[r], sh_r[r]);
      } else if (residB) {
        v += residB[(size_t)m * Tc + t];
      }
      Ob[(size_t)m * Tc + t] = v;
    }
  }
}

// ------------------------------------------- depthwise dilated conv + PReLU+BN
__global__ __launch_bounds__(256) void dwconv_kernel(
    const float* __restrict__ gin, float* __restrict__ out,
    const float* __restrict__ wd, const float* __restrict__ bd,
    const float* __restrict__ alpha_p,
    const float* __restrict__ bng, const float* __restrict__ bnb,
    const float* __restrict__ bnm, const float* __restrict__ bnv,
    int D, int Tc, int dil)
{
  int d = blockIdx.y;
  int b = blockIdx.z;
  const float* row = gin + ((size_t)b * D + d) * Tc;
  float*       orow = out + ((size_t)b * D + d) * Tc;
  float w0 = wd[d * 3 + 0], w1 = wd[d * 3 + 1], w2 = wd[d * 3 + 2];
  float bias = bd[d];
  float a = alpha_p[0];
  float scale = bng[d] * rsqrtf(bnv[d] + EPSBN);
  float shift = bnb[d] - bnm[d] * scale;
  int t = blockIdx.x * blockDim.x + threadIdx.x;
  if (t >= Tc) return;
  float xm = (t - dil >= 0) ? row[t - dil] : 0.f;
  float x0 = row[t];
  float xp = (t + dil < Tc) ? row[t + dil] : 0.f;
  float v = fmaf(xm, w0, fmaf(x0, w1, fmaf(xp, w2, bias)));
  v = (v > 0.f) ? v : a * v;
  orow[t] = fmaf(v, scale, shift);
}

// ---------------------------------------------------------- sigmoid mask * enc
__global__ __launch_bounds__(256) void mask_mul_kernel(
    const float* __restrict__ enc, const float* __restrict__ h,
    float* __restrict__ masked, size_t n)
{
  size_t i = (size_t)blockIdx.x * blockDim.x + threadIdx.x;
  if (i >= n) return;
  float s = 1.f / (1.f + expf(-h[i]));
  masked[i] = enc[i] * s;
}

// ------------------------------------------------- transposed-conv decoder
// out[b,s'] ; s = s'+FK ; taps: (t1 = s/STR, k1 = s%STR) and (t1-1, k1+STR)
__global__ __launch_bounds__(256) void decoder_kernel(
    const float* __restrict__ masked, const float* __restrict__ dw,
    const float* __restrict__ db, float* __restrict__ out,
    int E, int Tc, int T, int FK, int stride)
{
  int sp = blockIdx.x * blockDim.x + threadIdx.x;
  int b  = blockIdx.z;
  if (sp >= T) return;
  int s  = sp + FK;
  int t1 = s / stride;
  int k1 = s - t1 * stride;
  int t2 = t1 - 1;
  int k2 = k1 + stride;
  float acc = db[0];
  const float* mb = masked + (size_t)b * E * Tc;
  bool v1 = (t1 >= 0 && t1 < Tc);
  bool v2 = (t2 >= 0 && t2 < Tc);
  for (int e = 0; e < E; ++e) {
    const float* me = mb + (size_t)e * Tc;
    const float* we = dw + (size_t)e * FK;
    if (v1) acc = fmaf(me[t1], we[k1], acc);
    if (v2) acc = fmaf(me[t2], we[k2], acc);
  }
  out[(size_t)b * T + sp] = acc;
}

// ================================================================= launcher
extern "C" void kernel_launch(void* const* d_in, const int* in_sizes, int n_in,
                              void* d_out, int out_size, void* d_ws, size_t ws_size,
                              hipStream_t stream)
{
  (void)in_sizes; (void)n_in; (void)out_size; (void)ws_size;
  const float* x     = (const float*)d_in[0];
  const float* enc_w = (const float*)d_in[1];
  const float* enc_b = (const float*)d_in[2];
  const float* w1    = (const float*)d_in[3];
  const float* b1    = (const float*)d_in[4];
  const float* a1    = (const float*)d_in[5];
  const float* g1    = (const float*)d_in[6];
  const float* be1   = (const float*)d_in[7];
  const float* m1    = (const float*)d_in[8];
  const float* v1    = (const float*)d_in[9];
  const float* wd    = (const float*)d_in[10];
  const float* bd    = (const float*)d_in[11];
  const float* a2    = (const float*)d_in[12];
  const float* g2    = (const float*)d_in[13];
  const float* be2   = (const float*)d_in[14];
  const float* m2    = (const float*)d_in[15];
  const float* v2    = (const float*)d_in[16];
  const float* w2    = (const float*)d_in[17];
  const float* b2    = (const float*)d_in[18];
  const float* dec_w = (const float*)d_in[19];
  const float* dec_b = (const float*)d_in[20];
  float* out = (float*)d_out;

  const int B = 2, T = 64000, E = 256, D = 512, BL = 2, L = 6, FK = 20, STR = 10;
  const int Tc = (T + 2 * FK - FK) / STR + 1;   // 6403

  size_t encN = (size_t)B * E * Tc;
  size_t bigN = (size_t)B * D * Tc;
  float* enc    = (float*)d_ws;
  float* h      = enc + encN;
  float* resid  = h + encN;
  float* gbuf   = resid + encN;
  float* ubuf   = gbuf + bigN;
  float* masked = gbuf;   // reuse: g free by mask time

  // encoder
  {
    dim3 grid((Tc + 255) / 256, E, B);
    encoder_kernel<<<grid, 256, 0, stream>>>(x, enc_w, enc_b, enc, T, Tc, E, FK, STR, FK);
  }
  hipMemcpyAsync(h, enc, encN * sizeof(float), hipMemcpyDeviceToDevice, stream);

  dim3 ggrid1((Tc + NT - 1) / NT, D / MT, B);   // 1x1 conv: E->D
  dim3 ggrid2((Tc + NT - 1) / NT, E / MT, B);   // 1x1 conv: D->E
  dim3 dgrid((Tc + 255) / 256, D, B);

  for (int bI = 0; bI < BL; ++bI) {
    hipMemcpyAsync(resid, h, encN * sizeof(float), hipMemcpyDeviceToDevice, stream);
    for (int i = 0; i < L; ++i) {
      int li  = bI * L + i;
      int dil = 1 << i;
      // conv1 (E->D) + PReLU + BN
      gemm_fused_kernel<<<ggrid1, GTPB, 0, stream>>>(
          w1 + (size_t)li * D * E, h, gbuf, b1 + (size_t)li * D, nullptr,
          a1 + li, g1 + (size_t)li * D, be1 + (size_t)li * D,
          m1 + (size_t)li * D, v1 + (size_t)li * D,
          D, E, Tc, 0);
      // depthwise dilated conv + PReLU + BN
      dwconv_kernel<<<dgrid, 256, 0, stream>>>(
          gbuf, ubuf, wd + (size_t)li * D * 3, bd + (size_t)li * D, a2 + li,
          g2 + (size_t)li * D, be2 + (size_t)li * D,
          m2 + (size_t)li * D, v2 + (size_t)li * D,
          D, Tc, dil);
      // conv2 (D->E) + bias (+ residual on last layer of block)
      gemm_fused_kernel<<<ggrid2, GTPB, 0, stream>>>(
          w2 + (size_t)li * E * D, ubuf, h, b2 + (size_t)li * E,
          (i == L - 1) ? resid : nullptr,
          nullptr, nullptr, nullptr, nullptr, nullptr,
          E, D, Tc, 1);
    }
  }

  // mask * enc
  {
    int blocks = (int)((encN + 255) / 256);
    mask_mul_kernel<<<blocks, 256, 0, stream>>>(enc, h, masked, encN);
  }
  // decoder
  {
    dim3 grid((T + 255) / 256, 1, B);
    decoder_kernel<<<grid, 256, 0, stream>>>(masked, dec_w, dec_b, out, E, Tc, T, FK, STR);
  }
}
